// MSTransPoseNet_11690900979859
// MI455X (gfx1250) — matrix operand order repaired and checked by reading, compiled, not run on hardware
//
#include <hip/hip_runtime.h>
#include <hip/hip_bf16.h>
#include <math.h>

typedef __attribute__((ext_vector_type(16))) _Float16 v16h;
typedef __attribute__((ext_vector_type(8)))  _Float16 v8h;
typedef __attribute__((ext_vector_type(8)))  float    v8f;

#define B_    1024
#define S_    64
#define D_    256
#define CH_   1024
#define NT    64                 // hidden columns per block
#define A_STRIDE 264             // 256 + 8 f16 pad, keeps 16B alignment, spreads banks
#define B_STRIDE 272             // 256 + 16 f16 pad, keeps 16B alignment

__device__ __forceinline__ float gelu_exact(float x) {
  return 0.5f * x * (1.0f + erff(x * 0.70710678118654752f));
}

// ---------------------------------------------------------------------------
// Phase 1: scene logits, log-softmax, argmax, pose-bias seeding
// ---------------------------------------------------------------------------
__global__ __launch_bounds__(256)
void scene_kernel(const float* __restrict__ dt, const float* __restrict__ dr,
                  const float* __restrict__ Wsc, const float* __restrict__ bsc,
                  const float* __restrict__ bto, const float* __restrict__ bro,
                  float* __restrict__ out, int* __restrict__ max_idx)
{
  __shared__ float logits[S_];
  const int b = blockIdx.x;
  const int t = threadIdx.x;
  const int s = t >> 2;          // scene handled by this thread group
  const int p = t & 3;           // quarter of the 256-wide dot product

  const float4* vt = (const float4*)(dt + ((size_t)b * S_ + s) * D_ + p * 64);
  const float4* vr = (const float4*)(dr + ((size_t)b * S_ + s) * D_ + p * 64);
  const float4* wt = (const float4*)(Wsc + p * 64);
  const float4* wr = (const float4*)(Wsc + D_ + p * 64);

  float acc = 0.f;
#pragma unroll
  for (int i = 0; i < 16; ++i) {
    float4 a = vt[i], w = wt[i];
    acc += a.x * w.x + a.y * w.y + a.z * w.z + a.w * w.w;
    float4 c = vr[i], u = wr[i];
    acc += c.x * u.x + c.y * u.y + c.z * u.z + c.w * u.w;
  }
  acc += __shfl_xor(acc, 1, 32);
  acc += __shfl_xor(acc, 2, 32);
  if (p == 0) logits[s] = acc + bsc[0];
  __syncthreads();

  if (t < 32) {
    float l0 = logits[t], l1 = logits[t + 32];
    float m  = (l0 >= l1) ? l0 : l1;
    int   mi = (l0 >= l1) ? t  : t + 32;     // first-max tie break like argmax
#pragma unroll
    for (int off = 16; off; off >>= 1) {
      float om = __shfl_xor(m, off, 32);
      int   oi = __shfl_xor(mi, off, 32);
      if (om > m || (om == m && oi < mi)) { m = om; mi = oi; }
    }
    float se = expf(l0 - m) + expf(l1 - m);
#pragma unroll
    for (int off = 16; off; off >>= 1) se += __shfl_xor(se, off, 32);
    float logZ = m + logf(se);

    float* od = out + (size_t)B_ * 7 + (size_t)b * S_;   // scene_log_distr region
    od[t]      = l0 - logZ;
    od[t + 32] = l1 - logZ;

    if (t == 0) max_idx[b] = mi;
    float* po = out + (size_t)b * 7;                     // pose region: seed biases
    if (t < 3)      po[t] = bto[mi * 3 + t];
    else if (t < 7) po[t] = bro[mi * 4 + (t - 3)];
  }
}

// ---------------------------------------------------------------------------
// Phase 2a: counting-sort samples into per-scene buckets
// ---------------------------------------------------------------------------
__global__ __launch_bounds__(1024)
void bucket_kernel(const int* __restrict__ max_idx,
                   int* __restrict__ counts, int* __restrict__ bucket)
{
  __shared__ int lcnt[S_];
  const int t = threadIdx.x;
  if (t < S_) lcnt[t] = 0;
  __syncthreads();
  const int idx = max_idx[t];
  const int pos = atomicAdd(&lcnt[idx], 1);
  bucket[idx * B_ + pos] = t;
  __syncthreads();
  if (t < S_) counts[t] = lcnt[t];
}

// ---------------------------------------------------------------------------
// Phase 2b: per-scene expert GEMM (WMMA f16->f32) + GELU + output projection
// One block = (scene, 64-column tile). 4 waves, each owns a 16-column slice.
// ---------------------------------------------------------------------------
__global__ __launch_bounds__(128)
void expert_gemm_kernel(const float* __restrict__ descs,
                        const float* __restrict__ Wh,
                        const float* __restrict__ bh,
                        const float* __restrict__ Wo,
                        const int* __restrict__ counts,
                        const int* __restrict__ bucket,
                        float* __restrict__ out,
                        int n_out, int col0)
{
  __shared__ alignas(16) _Float16 Blds[NT * B_STRIDE];  // [col][K] column-major
  __shared__ alignas(16) _Float16 Alds[16 * A_STRIDE];  // [m][K]

  const int s   = blockIdx.x;
  const int cnt = counts[s];
  if (cnt == 0) return;
  const int nbase = blockIdx.y * NT;

  const int t    = threadIdx.x;
  const int lane = t & 31;
  const int wave = t >> 5;
  const int half = lane >> 4;
  const int nloc = lane & 15;

  // ---- convert f32 weight tile -> f16 column-major in LDS ----
  {
    const int ch = t & 63;
    const int d0 = t >> 6;                        // 0 or 1
    const float* wp = Wh + (size_t)s * D_ * CH_ + nbase + ch;
#pragma unroll 8
    for (int d = d0; d < D_; d += 2)
      Blds[ch * B_STRIDE + d] = (_Float16)wp[(size_t)d * CH_];
  }

  // per-lane column constants (hidden bias + output-weight column)
  const int col = nbase + wave * 16 + nloc;
  const float bias_h = bh[s * CH_ + col];
  float wo[4];
  for (int o = 0; o < n_out; ++o)
    wo[o] = Wo[((size_t)s * CH_ + col) * n_out + o];

  for (int c0 = 0; c0 < cnt; c0 += 16) {
    __syncthreads();   // B ready (first iter) / previous fragment reads done

    // ---- gather & convert up to 16 descriptor rows ----
    {
      const int i = t >> 3, j = t & 7;            // row, 32-element group
      _Float16* ap = Alds + i * A_STRIDE + j * 32;
      if (c0 + i < cnt) {
        const int smp = bucket[s * B_ + c0 + i];
        const float4* gp =
            (const float4*)(descs + ((size_t)smp * S_ + s) * D_ + j * 32);
#pragma unroll
        for (int e = 0; e < 8; ++e) {
          float4 v = gp[e];
          ap[e * 4 + 0] = (_Float16)v.x;  ap[e * 4 + 1] = (_Float16)v.y;
          ap[e * 4 + 2] = (_Float16)v.z;  ap[e * 4 + 3] = (_Float16)v.w;
        }
      } else {
#pragma unroll
        for (int e = 0; e < 32; ++e) ap[e] = (_Float16)0.f;
      }
    }
    __syncthreads();

    // ---- 16x16 tile per wave, K = 256 via 8 x wmma_f32_16x16x32_f16 ----
    v8f acc = {};
#pragma unroll
    for (int k = 0; k < 8; ++k) {
      const int kb = k * 32;
      union { v16h v; v8h h[2]; } a, bm;
      const _Float16* apx = Alds + nloc * A_STRIDE + kb + half * 8;
      a.h[0] = *(const v8h*)(apx);          // K = kb + half*8 .. +7
      a.h[1] = *(const v8h*)(apx + 16);     // K = kb + 16 + half*8 .. +7
      const _Float16* bpx = Blds + (wave * 16 + nloc) * B_STRIDE + kb + half * 8;
      bm.h[0] = *(const v8h*)(bpx);         // K = kb + half*8 .. +7
      bm.h[1] = *(const v8h*)(bpx + 16);    // K = kb + 16 + half*8 .. +7
      acc = __builtin_amdgcn_wmma_f32_16x16x32_f16(
          false, a.v, false, bm.v, (short)0, acc, false, false);
    }

    // ---- epilogue: hidden bias + exact GELU + output projection ----
#pragma unroll
    for (int r = 0; r < 8; ++r) {
      const int  m     = r + half * 8;               // C layout: VGPR r rows
      const bool valid = (c0 + m) < cnt;
      const float h    = gelu_exact(acc[r] + bias_h);
      for (int o = 0; o < n_out; ++o) {
        float v = h * wo[o];
#pragma unroll
        for (int off = 1; off < 16; off <<= 1)       // reduce over the 16 N lanes
          v += __shfl_xor(v, off, 32);
        if (nloc == 0 && valid) {
          const int smp = bucket[s * B_ + c0 + m];
          atomicAdd(out + (size_t)smp * 7 + col0 + o, v);
        }
      }
    }
  }
}

// ---------------------------------------------------------------------------
extern "C" void kernel_launch(void* const* d_in, const int* in_sizes, int n_in,
                              void* d_out, int out_size, void* d_ws, size_t ws_size,
                              hipStream_t stream)
{
  const float* dt  = (const float*)d_in[0];   // local_descs_t   [B,S,D]
  const float* dr  = (const float*)d_in[1];   // local_descs_rot [B,S,D]
  const float* Wsc = (const float*)d_in[2];   // W_scene [2D,1]
  const float* bsc = (const float*)d_in[3];   // b_scene [1]
  const float* Wth = (const float*)d_in[4];   // Wt_h [S,D,CH]
  const float* bth = (const float*)d_in[5];   // bt_h [S,CH]
  const float* Wto = (const float*)d_in[6];   // Wt_o [S,CH,3]
  const float* bto = (const float*)d_in[7];   // bt_o [S,3]
  const float* Wrh = (const float*)d_in[8];   // Wr_h [S,D,CH]
  const float* brh = (const float*)d_in[9];   // br_h [S,CH]
  const float* Wro = (const float*)d_in[10];  // Wr_o [S,CH,4]
  const float* bro = (const float*)d_in[11];  // br_o [S,4]
  float* out = (float*)d_out;                 // [B*7] pose | [B*S] log-softmax

  int* max_idx = (int*)d_ws;                  // [B]
  int* counts  = max_idx + B_;                // [S]
  int* bucket  = counts + S_;                 // [S][B]

  scene_kernel<<<B_, 256, 0, stream>>>(dt, dr, Wsc, bsc, bto, bro, out, max_idx);
  bucket_kernel<<<1, B_, 0, stream>>>(max_idx, counts, bucket);

  dim3 g(S_, CH_ / NT);
  expert_gemm_kernel<<<g, 128, 0, stream>>>(dt, Wth, bth, Wto, counts, bucket,
                                            out, 3, 0);
  expert_gemm_kernel<<<g, 128, 0, stream>>>(dr, Wrh, brh, Wro, counts, bucket,
                                            out, 4, 3);
}
